// SparseConvBlock_2207613190556
// MI455X (gfx1250) — matrix operand order repaired and checked by reading, not compiled, not run on hardware
//
#include <hip/hip_runtime.h>

#define DD 96
#define HH 320
#define WW 320

typedef __attribute__((ext_vector_type(16))) _Float16 v16h;
typedef __attribute__((ext_vector_type(8)))  _Float16 v8h;
typedef __attribute__((ext_vector_type(8)))  float    v8f;

static __device__ __forceinline__ v16h cat8(v8h a, v8h b) {
  return __builtin_shufflevector(a, b, 0,1,2,3,4,5,6,7,8,9,10,11,12,13,14,15);
}

#define WMMA_F16(A, B, C) \
  __builtin_amdgcn_wmma_f32_16x16x32_f16(false, (A), false, (B), (short)0, (C), false, false)

// ---------------- phase 0: init ----------------
__global__ void fill_grid_kernel(int* __restrict__ grid, int total) {
  int i = blockIdx.x * blockDim.x + threadIdx.x;
  if (i < total) grid[i] = -1;
}

__global__ void scatter_grid_kernel(const int* __restrict__ coords,
                                    int* __restrict__ grid, int n) {
  int i = blockIdx.x * blockDim.x + threadIdx.x;
  if (i >= n) return;
  int z = coords[3 * i + 0], y = coords[3 * i + 1], x = coords[3 * i + 2];
  grid[(z * HH + y) * WW + x] = i;
}

// ---------------- phase 1: precision conversion / weight repack ----------------
__global__ void cvt_feat_kernel(const float* __restrict__ src,
                                _Float16* __restrict__ dst, int total) {
  int i = blockIdx.x * blockDim.x + threadIdx.x;
  if (i < total) dst[i] = (_Float16)src[i];
}

// w16 layout: [27 taps][2 kchunks][4 nchunks][32 lanes][16 halves]
// B-fragment (K=32 x N=16, 16-bit): lanes 0-15 hold K {0..7,16..23},
// lanes 16-31 hold K {8..15,24..31}; N = lane%16.
__global__ void repack_w_kernel(const float* __restrict__ w,
                                _Float16* __restrict__ w16, int total) {
  int e = blockIdx.x * blockDim.x + threadIdx.x;
  if (e >= total) return;
  int h  = e & 15;
  int l  = (e >> 4) & 31;
  int nc = (e >> 9) & 3;
  int kc = (e >> 11) & 1;
  int t  = e >> 12;
  int K = kc * 32 + (h & 7) + ((h & 8) ? 16 : 0) + ((l & 16) ? 8 : 0);
  int N = nc * 16 + (l & 15);
  w16[e] = (_Float16)w[((size_t)t * 64 + K) * 64 + N];
}

// ---------------- phase 2: gather + WMMA conv ----------------
// One wave32 computes a 16x64 output tile: 4 accumulators (N-chunks of 16),
// K loop = 27 taps x 2 chunks of K=32.
__global__ __launch_bounds__(128) void conv_wmma_kernel(
    const _Float16* __restrict__ feat16,   // [n][64]
    const _Float16* __restrict__ w16,      // [27][2][4][32][16]
    const int* __restrict__ coords,        // [n][3]
    const int* __restrict__ grid,          // [DD][HH][WW]
    float* __restrict__ conv_out,          // [n][64]
    int n, int ntiles) {
  const int wave = threadIdx.x >> 5;
  const int lane = threadIdx.x & 31;
  const int tile = blockIdx.x * 4 + wave;
  if (tile >= ntiles) return;
  const int row_base = tile * 16;
  const int m = lane & 15;            // M (row within tile) / N (col within tile)
  const int hiHalf = lane >> 4;       // lane group selects K sub-range
  const int row = min(row_base + m, n - 1);

  const int cz = coords[3 * row + 0];
  const int cy = coords[3 * row + 1];
  const int cx = coords[3 * row + 2];

  v8f acc0 = {}, acc1 = {}, acc2 = {}, acc3 = {};
  const int base = hiHalf * 8;        // A-fragment per-lane K offset

  for (int dz = -1; dz <= 1; ++dz) {
    for (int dy = -1; dy <= 1; ++dy) {
      for (int dx = -1; dx <= 1; ++dx) {
        const int t = (dz + 1) * 9 + (dy + 1) * 3 + (dx + 1);
        const int nz = cz + dz, ny = cy + dy, nx = cx + dx;
        int idx = -1;
        if (nz >= 0 && nz < DD && ny >= 0 && ny < HH && nx >= 0 && nx < WW)
          idx = grid[(nz * HH + ny) * WW + nx];

        // wave-uniform skip: ~1.5% voxel density -> most off-center taps dead
        if (__ballot(idx >= 0) == 0ull) continue;

        v16h a0 = {}, a1 = {};
        if (idx >= 0) {
          const _Float16* rowp = feat16 + (size_t)idx * 64;
          // ISA 16-bit A 16x32 layout: lane lo holds K {0..7,16..23},
          // lane hi holds K {8..15,24..31} (per 32-wide K chunk).
          v8h x0 = *(const v8h*)(rowp + base);
          v8h x1 = *(const v8h*)(rowp + base + 16);
          v8h x2 = *(const v8h*)(rowp + 32 + base);
          v8h x3 = *(const v8h*)(rowp + 32 + base + 16);
          a0 = cat8(x0, x1);
          a1 = cat8(x2, x3);
        }

        const _Float16* wbase = w16 + (size_t)t * 4096 + (size_t)lane * 16;
        v16h b;
        b = *(const v16h*)(wbase + 0 * 512);        acc0 = WMMA_F16(a0, b, acc0);
        b = *(const v16h*)(wbase + 1 * 512);        acc1 = WMMA_F16(a0, b, acc1);
        b = *(const v16h*)(wbase + 2 * 512);        acc2 = WMMA_F16(a0, b, acc2);
        b = *(const v16h*)(wbase + 3 * 512);        acc3 = WMMA_F16(a0, b, acc3);
        b = *(const v16h*)(wbase + 2048 + 0 * 512); acc0 = WMMA_F16(a1, b, acc0);
        b = *(const v16h*)(wbase + 2048 + 1 * 512); acc1 = WMMA_F16(a1, b, acc1);
        b = *(const v16h*)(wbase + 2048 + 2 * 512); acc2 = WMMA_F16(a1, b, acc2);
        b = *(const v16h*)(wbase + 2048 + 3 * 512); acc3 = WMMA_F16(a1, b, acc3);
      }
    }
  }

  // D layout: VGPR v -> M = v + 8*hiHalf ; N = lane%16
#pragma unroll
  for (int v = 0; v < 8; ++v) {
    const int r = row_base + v + 8 * hiHalf;
    if (r < n) {
      float* op = conv_out + (size_t)r * 64;
      op[0 * 16 + m] = acc0[v];
      op[1 * 16 + m] = acc1[v];
      op[2 * 16 + m] = acc2[v];
      op[3 * 16 + m] = acc3[v];
    }
  }
}

// ---------------- phase 3: BatchNorm (deterministic) + LeakyReLU ----------------
__global__ void bn_partial_kernel(const float* __restrict__ x,
                                  float* __restrict__ part,  // [gridDim][128]
                                  int total) {
  __shared__ float s1[256];
  __shared__ float s2[256];
  float a = 0.f, b = 0.f;
  const int stride = gridDim.x * blockDim.x;     // multiple of 64
  for (int i = blockIdx.x * blockDim.x + threadIdx.x; i < total; i += stride) {
    float v = x[i];
    a += v;
    b += v * v;
  }
  s1[threadIdx.x] = a;
  s2[threadIdx.x] = b;
  __syncthreads();
  if (threadIdx.x < 64) {
    int t = threadIdx.x;
    a = s1[t] + s1[t + 64] + s1[t + 128] + s1[t + 192];
    b = s2[t] + s2[t + 64] + s2[t + 128] + s2[t + 192];
    part[(size_t)blockIdx.x * 128 + t] = a;
    part[(size_t)blockIdx.x * 128 + 64 + t] = b;
  }
}

__global__ void bn_finalize_kernel(const float* __restrict__ part,
                                   const float* __restrict__ gamma,
                                   const float* __restrict__ beta,
                                   float* __restrict__ sc,   // [128]: scale, shift
                                   int nblocks, int n) {
  int c = threadIdx.x;  // 64 threads
  float s = 0.f, q = 0.f;
  for (int b = 0; b < nblocks; ++b) {          // fixed order -> deterministic
    s += part[(size_t)b * 128 + c];
    q += part[(size_t)b * 128 + 64 + c];
  }
  float mean = s / (float)n;
  float var = q / (float)n - mean * mean;
  float inv = rsqrtf(var + 1e-5f);
  float sf = gamma[c] * inv;
  sc[c] = sf;
  sc[64 + c] = beta[c] - mean * sf;
}

__global__ void bn_apply_kernel(const float* __restrict__ x,
                                const float* __restrict__ sc,
                                float* __restrict__ out, int total) {
  int i = blockIdx.x * blockDim.x + threadIdx.x;
  if (i >= total) return;
  int c = i & 63;
  float y = x[i] * sc[c] + sc[64 + c];
  out[i] = (y >= 0.f) ? y : 0.01f * y;
}

// ---------------- launcher ----------------
extern "C" void kernel_launch(void* const* d_in, const int* in_sizes, int n_in,
                              void* d_out, int out_size, void* d_ws, size_t ws_size,
                              hipStream_t stream) {
  const float* features = (const float*)d_in[0];  // [n][64]
  const int*   coords   = (const int*)d_in[1];    // [n][3]
  const float* weight   = (const float*)d_in[2];  // [27][64][64]
  const float* gamma    = (const float*)d_in[3];  // [64]
  const float* beta     = (const float*)d_in[4];  // [64]
  float* out = (float*)d_out;

  const int n = in_sizes[0] / 64;
  const int grid_total = DD * HH * WW;
  const int w16_total = 27 * 2 * 4 * 32 * 16;     // 110592 halves
  const int NBS = 440;                            // BN partial blocks

  auto al = [](size_t x) { return (x + 255) & ~(size_t)255; };
  size_t off = 0;
  int*      g_grid  = (int*)((char*)d_ws + off);      off = al(off + (size_t)grid_total * 4);
  _Float16* feat16  = (_Float16*)((char*)d_ws + off); off = al(off + (size_t)n * 64 * 2);
  _Float16* w16     = (_Float16*)((char*)d_ws + off); off = al(off + (size_t)w16_total * 2);
  float*    convout = (float*)((char*)d_ws + off);    off = al(off + (size_t)n * 64 * 4);
  float*    part    = (float*)((char*)d_ws + off);    off = al(off + (size_t)NBS * 128 * 4);
  float*    sc      = (float*)((char*)d_ws + off);    off += 512;
  (void)ws_size; (void)n_in; (void)out_size;

  const int total_feat = n * 64;
  const int ntiles = (n + 15) / 16;

  fill_grid_kernel<<<(grid_total + 255) / 256, 256, 0, stream>>>(g_grid, grid_total);
  scatter_grid_kernel<<<(n + 255) / 256, 256, 0, stream>>>(coords, g_grid, n);
  cvt_feat_kernel<<<(total_feat + 255) / 256, 256, 0, stream>>>(features, feat16, total_feat);
  repack_w_kernel<<<(w16_total + 255) / 256, 256, 0, stream>>>(weight, w16, w16_total);

  conv_wmma_kernel<<<(ntiles + 3) / 4, 128, 0, stream>>>(
      feat16, w16, coords, g_grid, convout, n, ntiles);

  bn_partial_kernel<<<NBS, 256, 0, stream>>>(convout, part, total_feat);
  bn_finalize_kernel<<<1, 64, 0, stream>>>(part, gamma, beta, sc, NBS, n);
  bn_apply_kernel<<<(total_feat + 255) / 256, 256, 0, stream>>>(convout, sc, out, total_feat);
}
